// PointSIS_FollowMLP_28973849379049
// MI455X (gfx1250) — compile-verified, matrix operand order checked by reading, not run on hardware
//
// PointSIS forward for MI455X (gfx1250), compile-only tuning. Round 3.
//
// All dense layers run as bf16 WMMA GEMMs (v_wmma_f32_16x16x32_bf16, fp32
// accumulate). GEMM v2: block tile 256Mx64N, 8 waves x (32Mx64N) -> 8 WMMAs
// per 32-wide K-step per wave; B tile double-buffered in LDS via the CDNA5
// async global->LDS DMA path (ASYNCcnt) — builtin confirmed present, expects
// (int4 __device__*, int4 __shared__*, imm offset, imm cpol). A fragments
// stream directly from L2. Mamba selective scan = register-resident
// recurrence (3072 threads x 16 states, v_exp_f32 + FMA).
//
// Assumed d_in flatten order (setup_inputs insertion order, recursive):
//  0 s_feat (8192,16,3) 1 s_coord (8192,3) 2 s_order (2,8192) i32
//  3 s_inverse (2,8192) i32
//  4..7   pe: l1_w l1_b l2_w l2_b
//  8..19  fe: l1_w l1_b ln1_w ln1_b l2_w l2_b l3_w l3_b ln2_w ln2_b l4_w l4_b
//  20..25 fuse_e: l1_w l1_b ln_w ln_b l2_w l2_b
//  26..31 fuse_o: same
//  32 + m*24 + blk*11 + f  (m mixer, blk block):
//     f: 0 in_w 1 conv_w 2 conv_b 3 x_w 4 dt_w 5 dt_b 6 A_log 7 D 8 out_w
//        9 ln_w 10 ln_b ;  nf_w = 32+m*24+22, nf_b = +23.  n_in = 80.

#include <hip/hip_runtime.h>
#include <math.h>

typedef __bf16 bf16_t;
typedef __attribute__((ext_vector_type(16))) __bf16 v16bf;
typedef __attribute__((ext_vector_type(8))) float v8f;
typedef int v4i_t __attribute__((vector_size(16)));

#define DEV static __device__ __forceinline__

#if defined(__has_builtin)
#if __has_builtin(__builtin_amdgcn_global_load_async_to_lds_b128) && \
    __has_builtin(__builtin_amdgcn_s_wait_asynccnt)
#define USE_ASYNC_LDS 1
#endif
#endif
#ifndef USE_ASYNC_LDS
#define USE_ASYNC_LDS 0
#endif

DEV float bf2f(bf16_t h) {
  unsigned short u = __builtin_bit_cast(unsigned short, h);
  unsigned int x = ((unsigned int)u) << 16;
  return __builtin_bit_cast(float, x);
}
DEV bf16_t f2bf(float f) {
  unsigned int u = __builtin_bit_cast(unsigned int, f);
  unsigned int r = u + 0x7fffu + ((u >> 16) & 1u);
  unsigned short h = (unsigned short)(r >> 16);
  return __builtin_bit_cast(bf16_t, h);
}

// act: 0 none, 1 gelu(erf), 2 silu, 3 softplus
DEV float act_apply(float v, int act) {
  if (act == 1) return 0.5f * v * (1.0f + erff(v * 0.70710678118f));
  if (act == 2) return v / (1.0f + __expf(-v));
  if (act == 3) return (v > 20.0f) ? v : __logf(1.0f + __expf(v));
  return v;
}

#if USE_ASYNC_LDS
DEV void async_b128(const bf16_t* g, bf16_t* l) {
  __builtin_amdgcn_global_load_async_to_lds_b128(
      (__attribute__((address_space(1))) v4i_t*)(void*)g,
      (__attribute__((address_space(3))) v4i_t*)(void*)l, 0, 0);
}
#endif

// ---------------------------------------------------------------------------
// WMMA GEMM: C[M,N] = act( A[M,K](bf16) * W[N,K]^T(bf16) + bias + Cadd )
// Block tile 256(M) x 64(N); 8 waves, each owns 32x64 (2 A frags x 4 B frags
// = 8 accumulators). B tile (64x32) double-buffered in LDS, staged with the
// async global->LDS DMA when available; A fragments direct from global (L2).
// Requires M%256==0, N%64==0, K%32==0, ldw/lda % 8, 16B-aligned base ptrs.
// ---------------------------------------------------------------------------
__global__ __launch_bounds__(256) void k_gemm(
    const bf16_t* __restrict__ A, int lda, const bf16_t* __restrict__ W,
    int ldw, const float* __restrict__ bias, const float* __restrict__ Cadd,
    int caddGroup16, float* __restrict__ Cf, bf16_t* __restrict__ Cbf, int ldc,
    int M, int N, int K, int act) {
  __shared__ bf16_t Bs[2][64 * 32];
  const int tid = threadIdx.x;
  const int wave = tid >> 5;
  const int lane = tid & 31;
  const int kh = lane >> 4;   // K-half selector (CDNA5 16-bit A/B layout)
  const int l16 = lane & 15;
  const int gm0 = blockIdx.y * 256;
  const int gn0 = blockIdx.x * 64;
  const int tm0 = gm0 + wave * 32;

  // B-tile staging map: 256 threads x one b128 (8 bf16) = 64 rows x 32 cols
  const int srow = tid >> 2;
  const int scol = (tid & 3) * 8;
  const bf16_t* wstage = W + (size_t)(gn0 + srow) * ldw + scol;
  bf16_t* lstage0 = &Bs[0][srow * 32 + scol];
  bf16_t* lstage1 = &Bs[1][srow * 32 + scol];

  // A fragment pointers: lane(0-15)=row M, halves split K (0..15 | 16..31)
  const bf16_t* a0p = A + (size_t)(tm0 + l16) * lda + kh * 16;
  const bf16_t* a1p = A + (size_t)(tm0 + 16 + l16) * lda + kh * 16;

  v8f acc[2][4] = {};
  int p = 0;

#if USE_ASYNC_LDS
  async_b128(wstage, lstage0);
  for (int k0 = 0; k0 < K; k0 += 32) {
    __builtin_amdgcn_s_wait_asynccnt(0);
    __syncthreads();
    if (k0 + 32 < K) {
      async_b128(wstage + k0 + 32, p ? lstage0 : lstage1);
      __builtin_prefetch(a0p + k0 + 32, 0, 1);
    }
    v16bf a0 = *(const v16bf*)(a0p + k0);
    v16bf a1 = *(const v16bf*)(a1p + k0);
#pragma unroll
    for (int j = 0; j < 4; ++j) {
      v16bf b = *(const v16bf*)&Bs[p][(j * 16 + l16) * 32 + kh * 16];
      acc[0][j] = __builtin_amdgcn_wmma_f32_16x16x32_bf16(
          false, a0, false, b, (short)0, acc[0][j], false, false);
      acc[1][j] = __builtin_amdgcn_wmma_f32_16x16x32_bf16(
          false, a1, false, b, (short)0, acc[1][j], false, false);
    }
    p ^= 1;
  }
#else
  uint4 breg = *(const uint4*)wstage;
  for (int k0 = 0; k0 < K; k0 += 32) {
    *(uint4*)(p ? lstage1 : lstage0) = breg;
    __syncthreads();
    if (k0 + 32 < K) {
      breg = *(const uint4*)(wstage + k0 + 32);
      __builtin_prefetch(a0p + k0 + 32, 0, 1);
    }
    v16bf a0 = *(const v16bf*)(a0p + k0);
    v16bf a1 = *(const v16bf*)(a1p + k0);
#pragma unroll
    for (int j = 0; j < 4; ++j) {
      v16bf b = *(const v16bf*)&Bs[p][(j * 16 + l16) * 32 + kh * 16];
      acc[0][j] = __builtin_amdgcn_wmma_f32_16x16x32_bf16(
          false, a0, false, b, (short)0, acc[0][j], false, false);
      acc[1][j] = __builtin_amdgcn_wmma_f32_16x16x32_bf16(
          false, a1, false, b, (short)0, acc[1][j], false, false);
    }
    p ^= 1;
  }
#endif

#pragma unroll
  for (int t = 0; t < 2; ++t)
#pragma unroll
    for (int j = 0; j < 4; ++j)
#pragma unroll
      for (int i = 0; i < 8; ++i) {
        // C layout: VGPR i -> M = i (+8 for lanes 16-31), N = lane&15
        int r = tm0 + t * 16 + kh * 8 + i;
        int c = gn0 + j * 16 + l16;
        float v = acc[t][j][i];
        if (bias) v += bias[c];
        if (Cadd) {
          size_t ai = caddGroup16 ? ((size_t)(r >> 4) * ldc + c)
                                  : ((size_t)r * ldc + c);
          v += Cadd[ai];
        }
        v = act_apply(v, act);
        if (Cf) Cf[(size_t)r * ldc + c] = v;
        if (Cbf) Cbf[(size_t)r * ldc + c] = f2bf(v);
      }
}

// ---------------------------------------------------------------------------
// Row LayerNorm (+ optional activation). One block per row, cols <= 1024.
// ---------------------------------------------------------------------------
__global__ __launch_bounds__(256) void k_ln(
    const void* __restrict__ src, int srcIsBf, const float* __restrict__ w,
    const float* __restrict__ b, float* __restrict__ outF,
    bf16_t* __restrict__ outBf, int cols, int act) {
  const int row = blockIdx.x;
  const float* sf = (const float*)src;
  const bf16_t* sb = (const bf16_t*)src;
  __shared__ float red[256];
  float vals[4];
  int nv = 0;
  float s = 0.f;
  for (int c = threadIdx.x; c < cols; c += 256) {
    float v = srcIsBf ? bf2f(sb[(size_t)row * cols + c])
                      : sf[(size_t)row * cols + c];
    vals[nv++] = v;
    s += v;
  }
  red[threadIdx.x] = s;
  __syncthreads();
  for (int st = 128; st > 0; st >>= 1) {
    if (threadIdx.x < st) red[threadIdx.x] += red[threadIdx.x + st];
    __syncthreads();
  }
  float mean = red[0] / (float)cols;
  __syncthreads();
  s = 0.f;
  for (int i = 0; i < nv; ++i) {
    float d = vals[i] - mean;
    s += d * d;
  }
  red[threadIdx.x] = s;
  __syncthreads();
  for (int st = 128; st > 0; st >>= 1) {
    if (threadIdx.x < st) red[threadIdx.x] += red[threadIdx.x + st];
    __syncthreads();
  }
  float rstd = rsqrtf(red[0] / (float)cols + 1e-5f);
  int i = 0;
  for (int c = threadIdx.x; c < cols; c += 256) {
    float v = (vals[i++] - mean) * rstd * w[c] + b[c];
    v = act_apply(v, act);
    if (outF) outF[(size_t)row * cols + c] = v;
    if (outBf) outBf[(size_t)row * cols + c] = f2bf(v);
  }
}

// fp32 -> bf16 with zero padding (rows and cols)
__global__ void k_cvt_pad(const float* __restrict__ src,
                          bf16_t* __restrict__ dst, int srcR, int srcC,
                          int dstR, int dstC) {
  size_t i = (size_t)blockIdx.x * blockDim.x + threadIdx.x;
  size_t n = (size_t)dstR * dstC;
  if (i >= n) return;
  int r = (int)(i / dstC), c = (int)(i % dstC);
  float v = (r < srcR && c < srcC) ? src[(size_t)r * srcC + c] : 0.f;
  dst[i] = f2bf(v);
}

__global__ void k_negexp(const float* __restrict__ a, float* __restrict__ o,
                         int n) {
  int i = blockIdx.x * blockDim.x + threadIdx.x;
  if (i < n) o[i] = -__expf(a[i]);
}

// concat two f32 row blocks -> bf16 rows of width ca+cb
__global__ void k_concat2(const float* __restrict__ a, int ca,
                          const float* __restrict__ b, int cb,
                          bf16_t* __restrict__ out, int rows) {
  int cw = ca + cb;
  size_t i = (size_t)blockIdx.x * blockDim.x + threadIdx.x;
  if (i >= (size_t)rows * cw) return;
  int r = (int)(i / cw), c = (int)(i % cw);
  float v = (c < ca) ? a[(size_t)r * ca + c] : b[(size_t)r * cb + (c - ca)];
  out[i] = f2bf(v);
}

// per-group (16 rows) max over f32 -> bf16 (groups x cols)
__global__ void k_groupmax(const float* __restrict__ h,
                           bf16_t* __restrict__ g, int groups, int cols) {
  size_t i = (size_t)blockIdx.x * blockDim.x + threadIdx.x;
  if (i >= (size_t)groups * cols) return;
  int gi = (int)(i / cols), c = (int)(i % cols);
  float m = -3.4e38f;
  for (int r = 0; r < 16; ++r)
    m = fmaxf(m, h[((size_t)gi * 16 + r) * cols + c]);
  g[i] = f2bf(m);
}

// per-group (16 rows) max over f32 -> f32 (groups x cols)
__global__ void k_rowmax16(const float* __restrict__ t, float* __restrict__ o,
                           int groups, int cols) {
  size_t i = (size_t)blockIdx.x * blockDim.x + threadIdx.x;
  if (i >= (size_t)groups * cols) return;
  int gi = (int)(i / cols), c = (int)(i % cols);
  float m = -3.4e38f;
  for (int r = 0; r < 16; ++r)
    m = fmaxf(m, t[((size_t)gi * 16 + r) * cols + c]);
  o[i] = m;
}

// causal depthwise conv (k=4) + SiLU over (4 x 4096 x 768)
__global__ void k_conv4(const bf16_t* __restrict__ xs,
                        const float* __restrict__ cw,
                        const float* __restrict__ cb,
                        bf16_t* __restrict__ xc) {
  size_t i = (size_t)blockIdx.x * blockDim.x + threadIdx.x;
  if (i >= (size_t)16384 * 768) return;
  int d = (int)(i % 768);
  size_t t = i / 768;
  int l = (int)(t & 4095);
  float acc = cb[d];
#pragma unroll
  for (int k = 0; k < 4; ++k) {
    int li = l - 3 + k;
    if (li >= 0)
      acc += cw[d * 4 + k] * bf2f(xs[(t - (size_t)(3 - k)) * 768 + d]);
  }
  acc = acc / (1.f + __expf(-acc));  // silu
  xc[i] = f2bf(acc);
}

// Mamba selective scan: thread = (b, d), 16 states in registers, L=4096.
__global__ __launch_bounds__(256) void k_scan(
    const bf16_t* __restrict__ dt, const bf16_t* __restrict__ xc,
    const float* __restrict__ dbl, const float* __restrict__ Aneg,
    bf16_t* __restrict__ y) {
  int idx = blockIdx.x * blockDim.x + threadIdx.x;
  if (idx >= 4 * 768) return;
  int b = idx / 768, d = idx % 768;
  float a[16], h[16];
#pragma unroll
  for (int n = 0; n < 16; ++n) {
    a[n] = Aneg[d * 16 + n];
    h[n] = 0.f;
  }
  for (int l = 0; l < 4096; ++l) {
    size_t t = (size_t)b * 4096 + l;
    float dtv = bf2f(dt[t * 768 + d]);
    float xv = bf2f(xc[t * 768 + d]);
    float dx = dtv * xv;
    const float* p = dbl + t * 64;  // cols 24..39 = B, 40..55 = C
    float yv = 0.f;
#pragma unroll
    for (int n = 0; n < 16; ++n) {
      float dA = __expf(dtv * a[n]);
      h[n] = dA * h[n] + dx * p[24 + n];
      yv += h[n] * p[40 + n];
    }
    y[t * 768 + d] = f2bf(yv);
  }
}

// y = (y + xc*D[d]) * silu(z), in place on bf16 y
__global__ void k_gate(bf16_t* __restrict__ y, const bf16_t* __restrict__ xc,
                       const bf16_t* __restrict__ z,
                       const float* __restrict__ Dp, size_t n) {
  size_t i = (size_t)blockIdx.x * blockDim.x + threadIdx.x;
  if (i >= n) return;
  int d = (int)(i % 768);
  float v = bf2f(y[i]) + bf2f(xc[i]) * Dp[d];
  float zz = bf2f(z[i]);
  v *= zz / (1.f + __expf(-zz));
  y[i] = f2bf(v);
}

// seq[b, o*2048+g, :] = h[s_inverse[o][b*2048+g], :]   (f32 -> f32)
__global__ void k_gather_seq(const float* __restrict__ h,
                             const int* __restrict__ inv,
                             float* __restrict__ seq) {
  size_t i = (size_t)blockIdx.x * blockDim.x + threadIdx.x;
  if (i >= (size_t)16384 * 384) return;
  int c = (int)(i % 384);
  int row = (int)(i / 384);
  int b = row >> 12, r2 = row & 4095, o = r2 >> 11, g = r2 & 2047;
  int j = inv[o * 8192 + b * 2048 + g];
  seq[i] = h[(size_t)j * 384 + c];
}

// cat[j, o*384+c] = seqln[(b*4096 + o*2048 + g), c] with b*2048+g=s_order[o][j]
__global__ void k_seq_to_cat(const bf16_t* __restrict__ seqln,
                             const int* __restrict__ order,
                             bf16_t* __restrict__ cat) {
  size_t i = (size_t)blockIdx.x * blockDim.x + threadIdx.x;
  if (i >= (size_t)8192 * 768) return;
  int j = (int)(i / 768);
  int cc = (int)(i % 768);
  int o = cc / 384, c = cc % 384;
  int t = order[o * 8192 + j];
  int b = t >> 11, g = t & 2047;
  cat[i] = seqln[((size_t)(b * 4096 + o * 2048 + g)) * 384 + c];
}

// ---------------------------------------------------------------------------
extern "C" void kernel_launch(void* const* d_in, const int* in_sizes, int n_in,
                              void* d_out, int out_size, void* d_ws,
                              size_t ws_size, hipStream_t stream) {
  if (n_in < 80) return;
  if (ws_size < (size_t)300 * 1024 * 1024) return;

  const int TOK = 16384;  // B(4) * L(4096)

  const float* s_feat = (const float*)d_in[0];
  const float* s_coord = (const float*)d_in[1];
  const int* s_order = (const int*)d_in[2];
  const int* s_inv = (const int*)d_in[3];
  auto P = [&](int i) -> const float* { return (const float*)d_in[i]; };

  char* base = (char*)d_ws;
  size_t off = 0;
  auto alloc = [&](size_t bytes) -> void* {
    void* p = base + off;
    off = (off + bytes + 255) & ~(size_t)255;
    return p;
  };

  auto gemm = [&](const bf16_t* A, int lda, const bf16_t* W, int ldw,
                  const float* bias, const float* Cadd, int cg, float* Cf,
                  bf16_t* Cbf, int ldcv, int M, int N, int K, int act) {
    dim3 grid((unsigned)(N / 64), (unsigned)(M / 256));
    k_gemm<<<grid, dim3(256), 0, stream>>>(A, lda, W, ldw, bias, Cadd, cg, Cf,
                                           Cbf, ldcv, M, N, K, act);
  };
  auto ln = [&](const void* src, int isbf, const float* w, const float* b,
                float* of, bf16_t* ob, int rows, int cols, int act) {
    k_ln<<<rows, 256, 0, stream>>>(src, isbf, w, b, of, ob, cols, act);
  };
  auto cvt = [&](const float* src, bf16_t* dst, int sr, int sc, int dr,
                 int dc) {
    size_t n = (size_t)dr * dc;
    k_cvt_pad<<<(unsigned)((n + 255) / 256), 256, 0, stream>>>(src, dst, sr,
                                                               sc, dr, dc);
  };
  auto blocks = [](size_t n) { return (unsigned)((n + 255) / 256); };

  // ---- persistent: bf16 weights (padded) ----------------------------------
  bf16_t* pe1 = (bf16_t*)alloc(256 * 32 * 2);   cvt(P(4), pe1, 256, 3, 256, 32);
  bf16_t* pe2 = (bf16_t*)alloc(128 * 256 * 2);  cvt(P(6), pe2, 128, 256, 128, 256);
  bf16_t* fe1 = (bf16_t*)alloc(128 * 32 * 2);   cvt(P(8), fe1, 128, 3, 128, 32);
  bf16_t* fe2 = (bf16_t*)alloc(256 * 128 * 2);  cvt(P(12), fe2, 256, 128, 256, 128);
  bf16_t* fe3 = (bf16_t*)alloc(512 * 512 * 2);  cvt(P(14), fe3, 512, 512, 512, 512);
  bf16_t* fe4 = (bf16_t*)alloc(384 * 512 * 2);  cvt(P(18), fe4, 384, 512, 384, 512);
  bf16_t* fue1 = (bf16_t*)alloc(384 * 512 * 2); cvt(P(20), fue1, 384, 512, 384, 512);
  bf16_t* fue2 = (bf16_t*)alloc(384 * 384 * 2); cvt(P(24), fue2, 384, 384, 384, 384);
  bf16_t* fuo1 = (bf16_t*)alloc(384 * 768 * 2); cvt(P(26), fuo1, 384, 768, 384, 768);
  bf16_t* fuo2 = (bf16_t*)alloc(384 * 384 * 2); cvt(P(30), fuo2, 384, 384, 384, 384);

  bf16_t *inw[4], *xw[4], *dtw[4], *outw[4];
  float* Aneg[4];
  for (int m = 0; m < 2; ++m)
    for (int blk = 0; blk < 2; ++blk) {
      int bk = m * 2 + blk;
      int bi = 32 + m * 24 + blk * 11;
      inw[bk] = (bf16_t*)alloc((size_t)1536 * 384 * 2);
      cvt(P(bi + 0), inw[bk], 1536, 384, 1536, 384);
      xw[bk] = (bf16_t*)alloc(64 * 768 * 2);
      cvt(P(bi + 3), xw[bk], 56, 768, 64, 768);  // pad out-rows 56->64
      dtw[bk] = (bf16_t*)alloc(768 * 32 * 2);
      cvt(P(bi + 4), dtw[bk], 768, 24, 768, 32);  // pad K 24->32
      outw[bk] = (bf16_t*)alloc((size_t)384 * 768 * 2);
      cvt(P(bi + 8), outw[bk], 384, 768, 384, 768);
      Aneg[bk] = (float*)alloc(768 * 16 * 4);
      k_negexp<<<blocks(768 * 16), 256, 0, stream>>>(P(bi + 6), Aneg[bk],
                                                     768 * 16);
    }

  // ---- persistent activations ---------------------------------------------
  bf16_t* coordpad = (bf16_t*)alloc((size_t)8192 * 32 * 2);
  bf16_t* pebuf = (bf16_t*)alloc((size_t)8192 * 256 * 2);
  float* coordf = (float*)alloc((size_t)8192 * 128 * 4);
  float* hfeat = (float*)alloc((size_t)8192 * 384 * 4);
  float* rseq = (float*)alloc((size_t)TOK * 384 * 4);
  bf16_t* seqln = (bf16_t*)alloc((size_t)TOK * 384 * 2);

  // ---- position encoding: coord = l2(gelu(l1(s_coord))) -------------------
  cvt(s_coord, coordpad, 8192, 3, 8192, 32);
  gemm(coordpad, 32, pe1, 32, P(5), nullptr, 0, nullptr, pebuf, 256, 8192, 256,
       32, 1);
  gemm(pebuf, 256, pe2, 256, P(7), nullptr, 0, coordf, nullptr, 128, 8192, 128,
       256, 0);

  // ---- feature encoder, chunked 8x over the 131072 neighbor rows ----------
  {
    size_t ph = off;
    const int CR = 16384, CG = 1024;  // rows / groups per chunk
    bf16_t* fpad = (bf16_t*)alloc((size_t)CR * 32 * 2);
    bf16_t* t1b = (bf16_t*)alloc((size_t)CR * 128 * 2);
    bf16_t* t2b = (bf16_t*)alloc((size_t)CR * 128 * 2);
    float* h2f = (float*)alloc((size_t)CR * 256 * 4);
    bf16_t* h2b = (bf16_t*)alloc((size_t)CR * 256 * 2);
    bf16_t* gmx = (bf16_t*)alloc((size_t)CG * 256 * 2);
    float* gcf = (float*)alloc((size_t)CG * 512 * 4);
    bf16_t* t3b = (bf16_t*)alloc((size_t)CR * 512 * 2);
    bf16_t* t4b = (bf16_t*)alloc((size_t)CR * 512 * 2);
    float* t5f = (float*)alloc((size_t)CR * 384 * 4);
    for (int ch = 0; ch < 8; ++ch) {
      cvt(s_feat + (size_t)ch * CR * 3, fpad, CR, 3, CR, 32);
      gemm(fpad, 32, fe1, 32, P(9), nullptr, 0, nullptr, t1b, 128, CR, 128, 32,
           0);
      ln(t1b, 1, P(10), P(11), nullptr, t2b, CR, 128, 1);
      gemm(t2b, 128, fe2, 128, P(13), nullptr, 0, h2f, h2b, 256, CR, 256, 128,
           0);
      k_groupmax<<<blocks((size_t)CG * 256), 256, 0, stream>>>(h2f, gmx, CG,
                                                               256);
      // concat split: [h,g]@W3^T = h@W3[:, :256]^T + bcast(g@W3[:, 256:]^T)
      gemm(gmx, 256, fe3 + 256, 512, P(15), nullptr, 0, gcf, nullptr, 512, CG,
           512, 256, 0);
      gemm(h2b, 256, fe3, 512, nullptr, gcf, 1, nullptr, t3b, 512, CR, 512,
           256, 0);
      ln(t3b, 1, P(16), P(17), nullptr, t4b, CR, 512, 1);
      gemm(t4b, 512, fe4, 512, P(19), nullptr, 0, t5f, nullptr, 384, CR, 384,
           512, 0);
      k_rowmax16<<<blocks((size_t)CG * 384), 256, 0, stream>>>(
          t5f, hfeat + (size_t)ch * CG * 384, CG, 384);
    }
    off = ph;  // release FE scratch
  }

  // ---- mixer stages -------------------------------------------------------
  for (int s = 0; s < 2; ++s) {
    size_t ph = off;
    const float* hsrc = (s == 0) ? hfeat : (const float*)d_out;  // stage0 out

    bf16_t* catE = (bf16_t*)alloc((size_t)8192 * 512 * 2);
    bf16_t* t6 = (bf16_t*)alloc((size_t)8192 * 384 * 2);
    bf16_t* t7 = (bf16_t*)alloc((size_t)8192 * 384 * 2);
    float* hprime = (float*)alloc((size_t)8192 * 384 * 4);

    k_concat2<<<blocks((size_t)8192 * 512), 256, 0, stream>>>(hsrc, 384,
                                                              coordf, 128,
                                                              catE, 8192);
    gemm(catE, 512, fue1, 512, P(21), nullptr, 0, nullptr, t6, 384, 8192, 384,
         512, 0);
    ln(t6, 1, P(22), P(23), nullptr, t7, 8192, 384, 1);
    gemm(t7, 384, fue2, 384, P(25), nullptr, 0, hprime, nullptr, 384, 8192,
         384, 384, 0);
    k_gather_seq<<<blocks((size_t)TOK * 384), 256, 0, stream>>>(hprime, s_inv,
                                                                rseq);

    // mamba scratch
    bf16_t* xin = (bf16_t*)alloc((size_t)TOK * 384 * 2);
    bf16_t* xs = (bf16_t*)alloc((size_t)TOK * 768 * 2);
    bf16_t* zb = (bf16_t*)alloc((size_t)TOK * 768 * 2);
    bf16_t* xcb = (bf16_t*)alloc((size_t)TOK * 768 * 2);
    float* dblf = (float*)alloc((size_t)TOK * 64 * 4);
    bf16_t* dblb = (bf16_t*)alloc((size_t)TOK * 64 * 2);
    bf16_t* dtb = (bf16_t*)alloc((size_t)TOK * 768 * 2);
    bf16_t* yb = (bf16_t*)alloc((size_t)TOK * 768 * 2);

    for (int blk = 0; blk < 2; ++blk) {
      int bk = s * 2 + blk;
      int bi = 32 + s * 24 + blk * 11;
      ln(rseq, 0, P(bi + 9), P(bi + 10), nullptr, xin, TOK, 384, 0);
      // in_proj split into xs / z halves of in_w (1536x384)
      gemm(xin, 384, inw[bk], 384, nullptr, nullptr, 0, nullptr, xs, 768, TOK,
           768, 384, 0);
      gemm(xin, 384, inw[bk] + (size_t)768 * 384, 384, nullptr, nullptr, 0,
           nullptr, zb, 768, TOK, 768, 384, 0);
      k_conv4<<<blocks((size_t)TOK * 768), 256, 0, stream>>>(xs, P(bi + 1),
                                                             P(bi + 2), xcb);
      // x_proj (N padded 56->64): cols [0,24)=dt_raw [24,40)=B [40,56)=C
      gemm(xcb, 768, xw[bk], 768, nullptr, nullptr, 0, dblf, dblb, 64, TOK, 64,
           768, 0);
      // dt_proj (K padded 24->32, padded dt_w rows are zero) + softplus
      gemm(dblb, 64, dtw[bk], 32, P(bi + 5), nullptr, 0, nullptr, dtb, 768,
           TOK, 768, 32, 3);
      k_scan<<<12, 256, 0, stream>>>(dtb, xcb, dblf, Aneg[bk], yb);
      k_gate<<<blocks((size_t)TOK * 768), 256, 0, stream>>>(
          yb, xcb, zb, P(bi + 7), (size_t)TOK * 768);
      // out_proj with fused residual: rseq = rseq + y @ out_w^T
      gemm(yb, 768, outw[bk], 768, nullptr, rseq, 0, rseq, nullptr, 384, TOK,
           384, 768, 0);
    }
    int mb = 32 + s * 24;
    ln(rseq, 0, P(mb + 22), P(mb + 23), nullptr, seqln, TOK, 384, 0);

    bf16_t* catO = (bf16_t*)alloc((size_t)8192 * 768 * 2);
    bf16_t* t8 = (bf16_t*)alloc((size_t)8192 * 384 * 2);
    bf16_t* t9 = (bf16_t*)alloc((size_t)8192 * 384 * 2);
    k_seq_to_cat<<<blocks((size_t)8192 * 768), 256, 0, stream>>>(seqln,
                                                                 s_order, catO);
    gemm(catO, 768, fuo1, 768, P(27), nullptr, 0, nullptr, t8, 384, 8192, 384,
         768, 0);
    ln(t8, 1, P(28), P(29), nullptr, t9, 8192, 384, 1);
    gemm(t9, 384, fuo2, 384, P(31), nullptr, 0,
         (float*)d_out + (size_t)s * 8192 * 384, nullptr, 384, 8192, 384, 384,
         0);
    off = ph;  // release stage scratch
  }
}
